// TransformerGraphEmbeddingCosine_41291815584043
// MI455X (gfx1250) — matrix operand order repaired and checked by reading, not compiled, not run on hardware
//
#include <hip/hip_runtime.h>

// ---------------------------------------------------------------------------
// TransformerConv graph embedding + cosine, CDNA5 (gfx1250) WMMA + async-LDS
// ---------------------------------------------------------------------------

typedef __attribute__((ext_vector_type(16))) _Float16 v16h;
typedef __attribute__((ext_vector_type(8)))  float    v8f;

#define NN   10000     // nodes per graph
#define EE   160000    // edges per graph
#define DD   640       // hidden dim
#define EDD  8         // edge feature dim
#define BB   64        // graphs in batch
#define NL   5         // layers 1..5 (stacked weights)

union FragU { v16h v; uint4 u[2]; };

// CDNA5 async DMA: global -> LDS, tracked by ASYNCcnt (CDNA5 ISA)
__device__ __forceinline__ void async_copy_b128(unsigned lds_addr, unsigned long long gaddr) {
  asm volatile("global_load_async_to_lds_b128 %0, %1, off"
               :: "v"(lds_addr), "v"(gaddr) : "memory");
}
__device__ __forceinline__ void wait_async0() {
  asm volatile("s_wait_asynccnt 0x0" ::: "memory");
}

// ------------------------------ utility -----------------------------------

__global__ void zero_kernel(unsigned* __restrict__ p, int n) {
  int i = blockIdx.x * blockDim.x + threadIdx.x;
  if (i < n) p[i] = 0u;
}

// Pack fp32 weight W[K, DD] (row-major) into WMMA-B fragment layout, f16.
// Fragment = 32x16 (KxN) tile stored as 32 lanes x 16 halves:
//   lane<16 : col = nt*16+lane,    K = kt*32 + 0..15
//   lane>=16: col = nt*16+lane-16, K = kt*32 + 16..31
// Fragment order: [n_tile][k_tile]. Zero-pads K beyond Kin.
__global__ void pack_w_kernel(const float* __restrict__ W, _Float16* __restrict__ out,
                              int Kin, int Kp) {
  int t = blockIdx.x * blockDim.x + threadIdx.x;
  int total = DD * Kp;
  if (t >= total) return;
  int ktiles = Kp >> 5;
  int frag = t >> 9;           // /512 halves per fragment
  int r    = t & 511;
  int lane = r >> 4;
  int j    = r & 15;
  int n_tile = frag / ktiles;
  int k_tile = frag - n_tile * ktiles;
  int n = n_tile * 16 + (lane & 15);
  int k = k_tile * 32 + ((lane >> 4) << 4) + j;
  out[t] = (k < Kin) ? (_Float16)W[(size_t)k * DD + n] : (_Float16)0.0f;
}

// x [NN,4] fp32 -> f16 [NN,32] zero padded (layer-0 A matrix, K padded to 32)
__global__ void convert_x0_kernel(const float* __restrict__ x, _Float16* __restrict__ xh) {
  int t = blockIdx.x * blockDim.x + threadIdx.x;
  if (t >= NN * 32) return;
  int n = t >> 5, c = t & 31;
  xh[t] = (c < 4) ? (_Float16)x[n * 4 + c] : (_Float16)0.0f;
}

// relu + f32->f16 convert (next layer's A matrix)
__global__ void relu_convert_kernel(const float* __restrict__ in, _Float16* __restrict__ out,
                                    int n) {
  int i = blockIdx.x * blockDim.x + threadIdx.x;
  if (i < n) { float v = in[i]; out[i] = (_Float16)(v > 0.0f ? v : 0.0f); }
}

// ------------------------------ WMMA GEMM ----------------------------------
// C[M, DD] = A[M, Kp](f16 row-major) x Bpacked + bias.
// Block = 8 waves: 4 along M (16 rows) x 2 along N (64 cols) -> 64x128 tile.
// A tile + B fragments staged to LDS with async DMA, double buffered:
//   DMA(kt+1) overlaps ds_load+WMMA(kt); WMMA waits only on DScnt.

#define AS_STRIDE 40                 // halves per LDS A row (80 B, padded vs 64)
#define AS_BUF    (64 * AS_STRIDE)   // halves per A buffer
#define BS_BUF    (8 * 512)          // halves per B buffer (8 fragments)

__global__ __launch_bounds__(256)
void gemm_wmma_kernel(const _Float16* __restrict__ A,
                      const _Float16* __restrict__ Bp,
                      const float* __restrict__ bias,
                      float* __restrict__ C,
                      int M, int Kp) {
  __shared__ _Float16 As[2 * AS_BUF];
  __shared__ _Float16 Bs[2 * BS_BUF];

  const int tid  = threadIdx.x;
  const int lane = tid & 31;
  const int wave = tid >> 5;
  const int wm   = wave & 3;
  const int wn   = wave >> 2;
  const int m0   = blockIdx.x * 64;
  const int nt_base = blockIdx.y * 8;        // first 16-col tile of this block
  const int h    = lane >> 4;
  const int ktiles = Kp >> 5;

  // ---- per-thread async copy slots ----
  // A tile: 64 rows x 64 B -> 256 x b128 : thread copies row tid>>2, chunk tid&3
  int arow = m0 + (tid >> 2);
  if (arow >= M) arow = M - 1;               // clamp (stores masked later)
  const unsigned long long gA0 =
      (unsigned long long)(uintptr_t)(A + (size_t)arow * Kp + (tid & 3) * 8);
  const unsigned asBase = (unsigned)(uintptr_t)(&As[0]);
  const unsigned bsBase = (unsigned)(uintptr_t)(&Bs[0]);
  const unsigned ldsA0  = asBase + (unsigned)((tid >> 2) * 80 + (tid & 3) * 16);
  // B fragments: 8 x 1024 B -> 512 x b128 : two slots per thread
  const int f1_nt = tid >> 6,          f1_j = tid & 63;
  const int f2_nt = (tid + 256) >> 6,  f2_j = (tid + 256) & 63;
  const unsigned long long gBbase = (unsigned long long)(uintptr_t)Bp;

  auto issue = [&](int kt, int buf) {
    // A chunk
    async_copy_b128(ldsA0 + (unsigned)(buf * (AS_BUF * 2)),
                    gA0 + (unsigned long long)kt * 64ull);
    // B chunk 1
    {
      unsigned long long ga = gBbase +
          ((((unsigned long long)(nt_base + f1_nt)) * (unsigned)ktiles + (unsigned)kt) << 10) +
          (unsigned long long)(f1_j * 16);
      async_copy_b128(bsBase + (unsigned)(buf * (BS_BUF * 2) + f1_nt * 1024 + f1_j * 16), ga);
    }
    // B chunk 2
    {
      unsigned long long ga = gBbase +
          ((((unsigned long long)(nt_base + f2_nt)) * (unsigned)ktiles + (unsigned)kt) << 10) +
          (unsigned long long)(f2_j * 16);
      async_copy_b128(bsBase + (unsigned)(buf * (BS_BUF * 2) + f2_nt * 1024 + f2_j * 16), ga);
    }
  };

  v8f acc0 = {}, acc1 = {}, acc2 = {}, acc3 = {};

  issue(0, 0);
  for (int kt = 0; kt < ktiles; ++kt) {
    const int buf = kt & 1;
    wait_async0();        // this wave's DMA into buf complete
    __syncthreads();      // all waves' DMA complete; prior reads of buf^1 retired
    if (kt + 1 < ktiles) issue(kt + 1, buf ^ 1);   // overlap DMA with compute

    // A fragment (16-bit 16x32 layout) from LDS
    const _Float16* Abuf = As + buf * AS_BUF + (wm * 16 + (lane & 15)) * AS_STRIDE;
    FragU a;
    a.u[0] = *(const uint4*)(Abuf + h * 16);
    a.u[1] = *(const uint4*)(Abuf + 8 + h * 16);

    // 4 B fragments from LDS
    const _Float16* Bbuf = Bs + buf * BS_BUF + (wn * 4) * 512 + lane * 16;
    FragU b0, b1, b2, b3;
    b0.u[0] = *(const uint4*)(Bbuf);          b0.u[1] = *(const uint4*)(Bbuf + 8);
    b1.u[0] = *(const uint4*)(Bbuf + 512);    b1.u[1] = *(const uint4*)(Bbuf + 520);
    b2.u[0] = *(const uint4*)(Bbuf + 1024);   b2.u[1] = *(const uint4*)(Bbuf + 1032);
    b3.u[0] = *(const uint4*)(Bbuf + 1536);   b3.u[1] = *(const uint4*)(Bbuf + 1544);

    acc0 = __builtin_amdgcn_wmma_f32_16x16x32_f16(false, a.v, false, b0.v, (short)0, acc0, false, false);
    acc1 = __builtin_amdgcn_wmma_f32_16x16x32_f16(false, a.v, false, b1.v, (short)0, acc1, false, false);
    acc2 = __builtin_amdgcn_wmma_f32_16x16x32_f16(false, a.v, false, b2.v, (short)0, acc2, false, false);
    acc3 = __builtin_amdgcn_wmma_f32_16x16x32_f16(false, a.v, false, b3.v, (short)0, acc3, false, false);
  }

  // C/D layout: VGPR r, lanes0-15: M=r ; lanes16-31: M=8+r ; N = lane&15
  const int ncol  = lane & 15;
  const int rbase = m0 + wm * 16 + 8 * h;
  const int n0    = nt_base * 16 + wn * 64;
  v8f accs[4] = {acc0, acc1, acc2, acc3};
#pragma unroll
  for (int i = 0; i < 4; ++i) {
    const int col = n0 + i * 16 + ncol;
    const float bv = bias[col];
#pragma unroll
    for (int r = 0; r < 8; ++r) {
      const int row = rbase + r;
      if (row < M) C[(size_t)row * DD + col] = accs[i][r] + bv;
    }
  }
}

// ------------------------------ edge kernels -------------------------------
// One wave per edge. We (8x640) staged in LDS; edge embedding recomputed on
// the fly: e_d = sum_c ea[e,c] * We[c,d].

__global__ void edge_score_kernel(const int* __restrict__ ei, const float* __restrict__ ea,
                                  const float* __restrict__ We,
                                  const float* __restrict__ q, const float* __restrict__ k,
                                  float* __restrict__ score, unsigned* __restrict__ smax_u) {
  __shared__ float sWe[EDD * DD];
  for (int t = threadIdx.x; t < EDD * DD; t += blockDim.x) sWe[t] = We[t];
  __syncthreads();
  int e = blockIdx.x * 8 + (threadIdx.x >> 5);
  if (e >= EE) return;
  int lane = threadIdx.x & 31;
  int src = ei[e];
  int dst = ei[EE + e];
  float eav[EDD];
#pragma unroll
  for (int c = 0; c < EDD; ++c) eav[c] = ea[(size_t)e * EDD + c];
  float acc = 0.0f;
  for (int d = lane; d < DD; d += 32) {
    float ed = 0.0f;
#pragma unroll
    for (int c = 0; c < EDD; ++c) ed += eav[c] * sWe[c * DD + d];
    acc += q[(size_t)dst * DD + d] * (k[(size_t)src * DD + d] + ed);
  }
  for (int off = 16; off > 0; off >>= 1) acc += __shfl_xor(acc, off, 32);
  if (lane == 0) {
    float s = acc * 0.03952847075210474f;   // 1/sqrt(640)
    score[e] = s;
    unsigned bits = __float_as_uint(s);     // monotone float->uint for atomic max
    unsigned key = (bits & 0x80000000u) ? ~bits : (bits | 0x80000000u);
    atomicMax(&smax_u[dst], key);
  }
}

__global__ void edge_p_kernel(const int* __restrict__ ei, const float* __restrict__ score,
                              const unsigned* __restrict__ smax_u,
                              float* __restrict__ p, float* __restrict__ denom) {
  int e = blockIdx.x * blockDim.x + threadIdx.x;
  if (e >= EE) return;
  int dst = ei[EE + e];
  unsigned key  = smax_u[dst];
  unsigned bits = (key & 0x80000000u) ? (key ^ 0x80000000u) : ~key;
  float smax = __uint_as_float(bits);
  float pe = __expf(score[e] - smax);
  p[e] = pe;
  atomicAdd(&denom[dst], pe);
}

__global__ void edge_agg_kernel(const int* __restrict__ ei, const float* __restrict__ ea,
                                const float* __restrict__ We,
                                const float* __restrict__ v, const float* __restrict__ p,
                                const float* __restrict__ denom, float* __restrict__ out) {
  __shared__ float sWe[EDD * DD];
  for (int t = threadIdx.x; t < EDD * DD; t += blockDim.x) sWe[t] = We[t];
  __syncthreads();
  int e = blockIdx.x * 8 + (threadIdx.x >> 5);
  if (e >= EE) return;
  int lane = threadIdx.x & 31;
  int src = ei[e];
  int dst = ei[EE + e];
  float alpha = p[e] / (denom[dst] + 1e-16f);
  float eav[EDD];
#pragma unroll
  for (int c = 0; c < EDD; ++c) eav[c] = ea[(size_t)e * EDD + c];
  for (int d = lane; d < DD; d += 32) {
    float ed = 0.0f;
#pragma unroll
    for (int c = 0; c < EDD; ++c) ed += eav[c] * sWe[c * DD + d];
    atomicAdd(&out[(size_t)dst * DD + d], alpha * (v[(size_t)src * DD + d] + ed));
  }
}

// ------------------------------ head kernels -------------------------------

__global__ void pool_kernel(const float* __restrict__ h, const int* __restrict__ batch,
                            float* __restrict__ pooled) {
  int i = blockIdx.x * blockDim.x + threadIdx.x;
  if (i >= NN * DD) return;
  int n = i / DD, d = i - n * DD;
  atomicAdd(&pooled[batch[n] * DD + d], h[i]);
}

__global__ void ln_kernel(const float* __restrict__ pooled, const float* __restrict__ g,
                          const float* __restrict__ bta, float* __restrict__ z) {
  const int b = blockIdx.x;
  __shared__ float red[256];
  const float* row = pooled + b * DD;
  float s = 0.0f;
  for (int d = threadIdx.x; d < DD; d += 256) s += row[d];
  red[threadIdx.x] = s; __syncthreads();
  for (int o = 128; o > 0; o >>= 1) {
    if (threadIdx.x < o) red[threadIdx.x] += red[threadIdx.x + o];
    __syncthreads();
  }
  const float mu = red[0] * (1.0f / DD);
  __syncthreads();
  float s2 = 0.0f;
  for (int d = threadIdx.x; d < DD; d += 256) { float t = row[d] - mu; s2 += t * t; }
  red[threadIdx.x] = s2; __syncthreads();
  for (int o = 128; o > 0; o >>= 1) {
    if (threadIdx.x < o) red[threadIdx.x] += red[threadIdx.x + o];
    __syncthreads();
  }
  const float rstd = rsqrtf(red[0] * (1.0f / DD) + 1e-5f);
  for (int d = threadIdx.x; d < DD; d += 256)
    z[b * DD + d] = (row[d] - mu) * rstd * g[d] + bta[d];
}

__global__ void emb_kernel(const float* __restrict__ z, const float* __restrict__ Wemb,
                           const float* __restrict__ bemb, float* __restrict__ zemb) {
  const int b = blockIdx.x;
  __shared__ float sz[DD];
  for (int d = threadIdx.x; d < DD; d += 256) sz[d] = z[b * DD + d];
  __syncthreads();
  for (int o = threadIdx.x; o < DD; o += 256) {
    float acc = bemb[o];
    for (int kk = 0; kk < DD; ++kk) acc += sz[kk] * Wemb[kk * DD + o];
    zemb[b * DD + o] = acc > 0.0f ? acc : 0.0f;
  }
}

__global__ void cos_kernel(const float* __restrict__ zi, const float* __restrict__ zj,
                           float* __restrict__ outp) {
  const int b = blockIdx.x;
  __shared__ float r0[256], r1[256], r2[256];
  float dt = 0.0f, ni = 0.0f, nj = 0.0f;
  for (int d = threadIdx.x; d < DD; d += 256) {
    float a = zi[b * DD + d], c = zj[b * DD + d];
    dt += a * c; ni += a * a; nj += c * c;
  }
  r0[threadIdx.x] = dt; r1[threadIdx.x] = ni; r2[threadIdx.x] = nj; __syncthreads();
  for (int o = 128; o > 0; o >>= 1) {
    if (threadIdx.x < o) {
      r0[threadIdx.x] += r0[threadIdx.x + o];
      r1[threadIdx.x] += r1[threadIdx.x + o];
      r2[threadIdx.x] += r2[threadIdx.x + o];
    }
    __syncthreads();
  }
  if (threadIdx.x == 0) {
    float a = sqrtf(r1[0]); if (a < 1e-8f) a = 1e-8f;
    float c = sqrtf(r2[0]); if (c < 1e-8f) c = 1e-8f;
    outp[b] = r0[0] / (a * c);
  }
}

// ------------------------------ host driver --------------------------------

extern "C" void kernel_launch(void* const* d_in, const int* in_sizes, int n_in,
                              void* d_out, int out_size, void* d_ws, size_t ws_size,
                              hipStream_t stream) {
  const float* x_i     = (const float*)d_in[0];
  const int*   ei_i    = (const int*)  d_in[1];
  const float* ea_i    = (const float*)d_in[2];
  const int*   batch_i = (const int*)  d_in[3];
  const float* x_j     = (const float*)d_in[4];
  const int*   ei_j    = (const int*)  d_in[5];
  const float* ea_j    = (const float*)d_in[6];
  const int*   batch_j = (const int*)  d_in[7];
  const float* W0q = (const float*)d_in[8];  const float* b0q = (const float*)d_in[9];
  const float* W0k = (const float*)d_in[10]; const float* b0k = (const float*)d_in[11];
  const float* W0v = (const float*)d_in[12]; const float* b0v = (const float*)d_in[13];
  const float* W0e = (const float*)d_in[14];
  const float* W0s = (const float*)d_in[15]; const float* b0s = (const float*)d_in[16];
  const float* Wq  = (const float*)d_in[17]; const float* bq  = (const float*)d_in[18];
  const float* Wk  = (const float*)d_in[19]; const float* bk  = (const float*)d_in[20];
  const float* Wv  = (const float*)d_in[21]; const float* bv  = (const float*)d_in[22];
  const float* We  = (const float*)d_in[23];
  const float* Ws  = (const float*)d_in[24]; const float* bs  = (const float*)d_in[25];
  const float* ln_g = (const float*)d_in[26]; const float* ln_b = (const float*)d_in[27];
  const float* Wemb = (const float*)d_in[28]; const float* bemb = (const float*)d_in[29];
  float* outp = (float*)d_out;

  // workspace bump allocator
  char* wsp = (char*)d_ws;
  size_t off = 0;
  auto alloc = [&](size_t bytes) -> void* {
    void* pp = wsp + off;
    off += (bytes + 255) & ~(size_t)255;
    return pp;
  };

  _Float16* Wp0q = (_Float16*)alloc((size_t)32 * DD * 2);
  _Float16* Wp0k = (_Float16*)alloc((size_t)32 * DD * 2);
  _Float16* Wp0v = (_Float16*)alloc((size_t)32 * DD * 2);
  _Float16* Wp0s = (_Float16*)alloc((size_t)32 * DD * 2);
  _Float16* WpL  = (_Float16*)alloc((size_t)NL * 4 * DD * DD * 2);
  _Float16* x0h  = (_Float16*)alloc((size_t)NN * 32 * 2);
  _Float16* Hh   = (_Float16*)alloc((size_t)NN * DD * 2);
  float* qb    = (float*)alloc((size_t)NN * DD * 4);
  float* kb    = (float*)alloc((size_t)NN * DD * 4);
  float* vb    = (float*)alloc((size_t)NN * DD * 4);
  float* ob    = (float*)alloc((size_t)NN * DD * 4);
  float* score = (float*)alloc((size_t)EE * 4);
  float* pvec  = (float*)alloc((size_t)EE * 4);
  unsigned* smax_u = (unsigned*)alloc((size_t)NN * 4);
  float* denom  = (float*)alloc((size_t)NN * 4);
  float* pooled = (float*)alloc((size_t)BB * DD * 4);
  float* zbuf   = (float*)alloc((size_t)BB * DD * 4);
  float* zemb_i = (float*)alloc((size_t)BB * DD * 4);
  float* zemb_j = (float*)alloc((size_t)BB * DD * 4);

  auto WpAt = [&](int l, int m) { return WpL + (size_t)(l * 4 + m) * DD * DD; };

  // ---- pack weights into WMMA B-fragment layout (once per call) ----
  auto packW = [&](const float* W, _Float16* dst, int Kin, int Kp) {
    int total = DD * Kp;
    pack_w_kernel<<<(total + 255) / 256, 256, 0, stream>>>(W, dst, Kin, Kp);
  };
  packW(W0q, Wp0q, 4, 32);
  packW(W0k, Wp0k, 4, 32);
  packW(W0v, Wp0v, 4, 32);
  packW(W0s, Wp0s, 4, 32);
  for (int l = 0; l < NL; ++l) {
    packW(Wq + (size_t)l * DD * DD, WpAt(l, 0), DD, DD);
    packW(Wk + (size_t)l * DD * DD, WpAt(l, 1), DD, DD);
    packW(Wv + (size_t)l * DD * DD, WpAt(l, 2), DD, DD);
    packW(Ws + (size_t)l * DD * DD, WpAt(l, 3), DD, DD);
  }

  auto gemm = [&](const _Float16* A, const _Float16* Bp, const float* bias, float* Cc, int Kp) {
    dim3 g((NN + 63) / 64, DD / 128);
    gemm_wmma_kernel<<<g, 256, 0, stream>>>(A, Bp, bias, Cc, NN, Kp);
  };

  auto edge_pass = [&](const int* ei, const float* ea, const float* WeL) {
    zero_kernel<<<(NN + 255) / 256, 256, 0, stream>>>(smax_u, NN);
    zero_kernel<<<(NN + 255) / 256, 256, 0, stream>>>((unsigned*)denom, NN);
    edge_score_kernel<<<(EE + 7) / 8, 256, 0, stream>>>(ei, ea, WeL, qb, kb, score, smax_u);
    edge_p_kernel<<<(EE + 255) / 256, 256, 0, stream>>>(ei, score, smax_u, pvec, denom);
    edge_agg_kernel<<<(EE + 7) / 8, 256, 0, stream>>>(ei, ea, WeL, vb, pvec, denom, ob);
  };

  auto run_graph = [&](const float* x, const int* ei, const float* ea, const int* batch,
                       float* zemb) {
    // layer 0 (K padded 4 -> 32)
    convert_x0_kernel<<<(NN * 32 + 255) / 256, 256, 0, stream>>>(x, x0h);
    gemm(x0h, Wp0q, b0q, qb, 32);
    gemm(x0h, Wp0k, b0k, kb, 32);
    gemm(x0h, Wp0v, b0v, vb, 32);
    gemm(x0h, Wp0s, b0s, ob, 32);     // skip connection pre-written into ob
    edge_pass(ei, ea, W0e);           // aggregation atomically added into ob
    relu_convert_kernel<<<(NN * DD + 255) / 256, 256, 0, stream>>>(ob, Hh, NN * DD);

    // layers 1..5
    for (int l = 0; l < NL; ++l) {
      gemm(Hh, WpAt(l, 0), bq + l * DD, qb, DD);
      gemm(Hh, WpAt(l, 1), bk + l * DD, kb, DD);
      gemm(Hh, WpAt(l, 2), bv + l * DD, vb, DD);
      gemm(Hh, WpAt(l, 3), bs + l * DD, ob, DD);
      edge_pass(ei, ea, We + (size_t)l * EDD * DD);
      if (l < NL - 1)
        relu_convert_kernel<<<(NN * DD + 255) / 256, 256, 0, stream>>>(ob, Hh, NN * DD);
    }

    // pool -> LN -> dense head
    zero_kernel<<<(BB * DD + 255) / 256, 256, 0, stream>>>((unsigned*)pooled, BB * DD);
    pool_kernel<<<(NN * DD + 255) / 256, 256, 0, stream>>>(ob, batch, pooled);
    ln_kernel<<<BB, 256, 0, stream>>>(pooled, ln_g, ln_b, zbuf);
    emb_kernel<<<BB, 256, 0, stream>>>(zbuf, Wemb, bemb, zemb);
  };

  run_graph(x_i, ei_i, ea_i, batch_i, zemb_i);
  run_graph(x_j, ei_j, ea_j, batch_j, zemb_j);
  cos_kernel<<<BB, 256, 0, stream>>>(zemb_i, zemb_j, outp);
}
